// InfinityMambaWithMiras_51565377356267
// MI455X (gfx1250) — matrix operand order repaired and checked by reading, not compiled, not run on hardware
//
#include <hip/hip_runtime.h>
#include <hip/hip_bf16.h>
#include <math.h>

// Problem sizes (from reference)
#define B_    8
#define T_    512
#define D_    1024
#define S_    2048
#define H_    4
#define Dh_   256
#define TOPK_ 8

typedef __bf16 bf16;
typedef __attribute__((ext_vector_type(16))) __bf16 v16bf;
typedef __attribute__((ext_vector_type(8)))  float  v8f;
typedef __attribute__((ext_vector_type(4)))  unsigned v4u_t;
typedef __attribute__((ext_vector_type(8)))  int      v8i_t;
typedef __attribute__((ext_vector_type(4)))  int      v4i_t;

#ifndef __has_builtin
#define __has_builtin(x) 0
#endif
#if __has_builtin(__builtin_amdgcn_tensor_load_to_lds)
#define HAVE_TDM 1
#else
#define HAVE_TDM 0
#endif

__device__ __forceinline__ bf16 f2bf(float f) {
  union { float f; unsigned u; } x; x.f = f;
  unsigned r = x.u + 0x7fffu + ((x.u >> 16) & 1u);   // round-to-nearest-even
  unsigned short hs = (unsigned short)(r >> 16);
  union { unsigned short s; bf16 b; } y; y.s = hs;
  return y.b;
}

union Frag { uint4 u[2]; v16bf v; };

// LDS byte offset of a generic pointer into shared memory:
// per CDNA5 ISA 10.2, the LDS offset is the low 32 bits of the flat address.
__device__ __forceinline__ unsigned lds_offset(const void* p) {
  return (unsigned)(size_t)p;
}

#if HAVE_TDM
// ---------------------------------------------------------------------------
// Issue a TDM load of a 64x64 bf16 tile (row length K elems in global) into
// LDS with a 144-byte row stride (pad_interval=32 dwords -> +4 dwords pad).
// D# layout per CDNA5 ISA 8.3/8.4 (group0 128b, group1 256b; groups 2/3 zero
// for a 2-D tensor). Wave-level op, tracked by TENSORcnt.
// This toolchain exposes the 6-arg builtin (extra int32x8 group slot).
// ---------------------------------------------------------------------------
__device__ __forceinline__ void tdm_load_tile64x64(const bf16* gsrc, unsigned ldsOff, int K) {
  unsigned long long ga = (unsigned long long)(size_t)gsrc;
  v4u_t g0;
  g0[0] = 1u;                                                // count=1 (user D#)
  g0[1] = ldsOff;                                            // lds_addr (bytes)
  g0[2] = (unsigned)ga;                                      // global_addr[31:0]
  g0[3] = (unsigned)((ga >> 32) & 0x01ffffffull) | (2u << 30); // addr[56:32] | type=2
  v8i_t g1;
  // data_size=1 (2B), pad_enable, pad_interval code 4 (32 dwords), pad_amount code 3 (4 dwords)
  unsigned dw0 = (1u << 16) | (1u << 20) | (4u << 22) | (3u << 25);
  unsigned td0 = (unsigned)K;        // tensor_dim0 = row length (elems)
  unsigned td1 = 0x100000u;          // tensor_dim1: generous (OOB never hit)
  g1[0] = (int)dw0;
  g1[1] = (int)((td0 & 0xffffu) << 16);                      // tensor_dim0[15:0]
  g1[2] = (int)((td0 >> 16) | ((td1 & 0xffffu) << 16));      // dim0[31:16] | dim1[15:0]
  g1[3] = (int)((td1 >> 16) | (64u << 16));                  // dim1[31:16] | tile_dim0=64
  g1[4] = (int)64u;                                          // tile_dim1=64, tile_dim2=0
  g1[5] = (int)(unsigned)K;                                  // tensor_dim0_stride lo
  g1[6] = 0;                                                 // stride0 hi | stride1 lo
  g1[7] = 0;                                                 // stride1 hi
  v4i_t z4 = {0, 0, 0, 0};
  v8i_t z8 = {0, 0, 0, 0, 0, 0, 0, 0};
  __builtin_amdgcn_tensor_load_to_lds(g0, g1, z4, z4, z8, 0);
}

// ---------------------------------------------------------------------------
// Backbone WMMA GEMM, TDM-staged: C[M,N] = A_bf16[M,K] @ Bt_bf16[N,K]^T.
// Block tile 64x64, K-step 64, double-buffered LDS filled by TDM (wave 0
// issues, s_wait_tensorcnt + barrier synchronizes). 8 waves in a 2(M)x4(N)
// grid; each wave owns two 16x16 M-tiles -> 4 v_wmma per K-step.
// EPI==0: GELU -> bf16 out.  EPI==1: +bias -> f32 out.
// ---------------------------------------------------------------------------
template<int EPI>
__global__ __launch_bounds__(256) void wmma_gemm_kernel(
    const bf16* __restrict__ A,     // [M][K]
    const bf16* __restrict__ Bt,    // [N][K]
    const float* __restrict__ bias, // [N]
    void* __restrict__ out, int M, int N, int K)
{
  __shared__ __align__(16) bf16 sA[2][64 * 72];   // 144B row stride (TDM pad)
  __shared__ __align__(16) bf16 sB[2][64 * 72];
  const int tid  = threadIdx.x;
  const int wave = tid >> 5;
  const int lane = tid & 31;
  const int wm = wave >> 2, wn = wave & 3;
  const int hi = lane >> 4;
  const int row0 = blockIdx.y * 64;
  const int col0 = blockIdx.x * 64;
  const int nK = K >> 6;

  if (tid < 32) {  // wave 0: prologue fill of buffer 0
    tdm_load_tile64x64(A  + (size_t)row0 * K, lds_offset(&sA[0][0]), K);
    tdm_load_tile64x64(Bt + (size_t)col0 * K, lds_offset(&sB[0][0]), K);
  }

  v8f acc[2] = { v8f{}, v8f{} };
  for (int kt = 0; kt < nK; ++kt) {
    const int cur = kt & 1;
    if (tid < 32) {
      if (kt + 1 < nK) {
        tdm_load_tile64x64(A  + (size_t)row0 * K + (size_t)(kt + 1) * 64,
                           lds_offset(&sA[cur ^ 1][0]), K);
        tdm_load_tile64x64(Bt + (size_t)col0 * K + (size_t)(kt + 1) * 64,
                           lds_offset(&sB[cur ^ 1][0]), K);
        __builtin_amdgcn_s_wait_tensorcnt(2);   // current pair complete (in-order)
      } else {
        __builtin_amdgcn_s_wait_tensorcnt(0);
      }
    }
    __syncthreads();                            // tile `cur` visible to all waves

    #pragma unroll
    for (int ks = 0; ks < 2; ++ks) {            // two 32-wide K sub-steps
      Frag fb;
      const bf16* bp = &sB[cur][(wn * 16 + (lane & 15)) * 72 + ks * 32 + hi * 16];
      fb.u[0] = *(const uint4*)bp;              // B: K = kb..kb+15 contiguous
      fb.u[1] = *(const uint4*)(bp + 8);
      #pragma unroll
      for (int tr = 0; tr < 2; ++tr) {
        Frag fa;
        const bf16* ap = &sA[cur][(wm * 32 + tr * 16 + (lane & 15)) * 72 + ks * 32 + hi * 16];
        fa.u[0] = *(const uint4*)ap;            // A: K = kb..kb+7
        fa.u[1] = *(const uint4*)(ap + 8);      //    K = kb+8..kb+15
        acc[tr] = __builtin_amdgcn_wmma_f32_16x16x32_bf16(
            false, fa.v, false, fb.v, (short)0, acc[tr], false, false);
      }
    }
    __syncthreads();                            // done reading before overwrite
  }

  const int gn = col0 + wn * 16 + (lane & 15);
  const float bn = bias[gn];
  #pragma unroll
  for (int tr = 0; tr < 2; ++tr) {
    #pragma unroll
    for (int r = 0; r < 8; ++r) {
      const int gm = row0 + wm * 32 + tr * 16 + r + hi * 8;
      float v = acc[tr][r] + bn;
      if (EPI == 0) {
        float tv = 0.7978845608028654f * (v + 0.044715f * v * v * v);
        float g  = 0.5f * v * (1.0f + tanhf(tv));
        ((bf16*)out)[(size_t)gm * N + gn] = f2bf(g);
      } else {
        ((float*)out)[(size_t)gm * N + gn] = v;
      }
    }
  }
}
#else
// ---------------------------------------------------------------------------
// Fallback (no TDM builtin): direct-from-global WMMA GEMM.
// ---------------------------------------------------------------------------
template<int EPI>
__global__ __launch_bounds__(256) void wmma_gemm_kernel(
    const bf16* __restrict__ A, const bf16* __restrict__ Bt,
    const float* __restrict__ bias, void* __restrict__ out,
    int M, int N, int K)
{
  const int wave = threadIdx.x >> 5;
  const int lane = threadIdx.x & 31;
  const int wm = wave >> 2, wn = wave & 3;
  const int row0 = blockIdx.y * 32 + wm * 16;
  const int col0 = blockIdx.x * 64 + wn * 16;
  const int hi   = lane >> 4;
  const bf16* aptr = A  + (size_t)(row0 + (lane & 15)) * K + hi * 16;
  const bf16* bptr = Bt + (size_t)(col0 + (lane & 15)) * K + hi * 16;
  v8f acc = {};
  for (int k = 0; k < K; k += 32) {
    Frag fa, fb;
    fa.u[0] = *(const uint4*)(aptr + k);
    fa.u[1] = *(const uint4*)(aptr + k + 8);
    fb.u[0] = *(const uint4*)(bptr + k);
    fb.u[1] = *(const uint4*)(bptr + k + 8);
    __builtin_prefetch(aptr + k + 128, 0, 1);
    __builtin_prefetch(bptr + k + 128, 0, 1);
    acc = __builtin_amdgcn_wmma_f32_16x16x32_bf16(false, fa.v, false, fb.v,
                                                  (short)0, acc, false, false);
  }
  const int gn = col0 + (lane & 15);
  const float bn = bias[gn];
  #pragma unroll
  for (int r = 0; r < 8; ++r) {
    const int gm = row0 + r + hi * 8;
    float v = acc[r] + bn;
    if (EPI == 0) {
      float tv = 0.7978845608028654f * (v + 0.044715f * v * v * v);
      float g  = 0.5f * v * (1.0f + tanhf(tv));
      ((bf16*)out)[(size_t)gm * N + gn] = f2bf(g);
    } else {
      ((float*)out)[(size_t)gm * N + gn] = v;
    }
  }
}
#endif

// ---------------------------------------------------------------------------
// f32 -> bf16 cast (activations) and cast+transpose (weights [K,N] -> [N,K])
// ---------------------------------------------------------------------------
__global__ void cast_kernel(const float* __restrict__ src, bf16* __restrict__ dst, size_t n) {
  size_t i = (size_t)blockIdx.x * 256 + threadIdx.x;
  if (i < n) dst[i] = f2bf(src[i]);
}

__global__ void cast_transpose_kernel(const float* __restrict__ src, bf16* __restrict__ dst,
                                      int K, int N) {
  size_t idx = (size_t)blockIdx.x * 256 + threadIdx.x;
  if (idx >= (size_t)K * N) return;
  int k = (int)(idx / N), n = (int)(idx % N);
  dst[(size_t)n * K + k] = f2bf(src[idx]);
}

// ---------------------------------------------------------------------------
// Backbone residual LayerNorm: h += LN(y); also emit bf16 copy of new h.
// ---------------------------------------------------------------------------
__global__ __launch_bounds__(256) void lnres_kernel(
    const float* __restrict__ y, const float* __restrict__ g, const float* __restrict__ be,
    float* __restrict__ h, bf16* __restrict__ hbf)
{
  const int row = blockIdx.x;
  const int tid = threadIdx.x;
  __shared__ float red[256];
  __shared__ float m_s, v_s;
  float s1 = 0.f, s2 = 0.f;
  for (int d = tid; d < D_; d += 256) {
    float v = y[(size_t)row * D_ + d];
    s1 += v; s2 += v * v;
  }
  red[tid] = s1; __syncthreads();
  for (int off = 128; off > 0; off >>= 1) { if (tid < off) red[tid] += red[tid + off]; __syncthreads(); }
  if (tid == 0) m_s = red[0] * (1.0f / D_);
  __syncthreads();
  red[tid] = s2; __syncthreads();
  for (int off = 128; off > 0; off >>= 1) { if (tid < off) red[tid] += red[tid + off]; __syncthreads(); }
  if (tid == 0) v_s = red[0] * (1.0f / D_) - m_s * m_s;
  __syncthreads();
  const float rs = rsqrtf(v_s + 1e-5f);
  for (int d = tid; d < D_; d += 256) {
    float v  = y[(size_t)row * D_ + d];
    float o  = (v - m_s) * rs * g[d] + be[d];
    float hn = h[(size_t)row * D_ + d] + o;
    h[(size_t)row * D_ + d]   = hn;
    hbf[(size_t)row * D_ + d] = f2bf(hn);
  }
}

// ---------------------------------------------------------------------------
// Scan step kernels (launched once per t, stream-ordered)
// ---------------------------------------------------------------------------

// scores[b][h][s] = (q_bh . K_sh) / sqrt(Dh).  One wave per slot s.
__global__ __launch_bounds__(256) void scores_kernel(
    const float* __restrict__ h, const float* __restrict__ mK,
    float* __restrict__ scores, int t)
{
  const int wave = threadIdx.x >> 5;
  const int lane = threadIdx.x & 31;
  const int s = blockIdx.x * 8 + wave;
  const float* krow = mK + (size_t)s * D_ + lane * 32;
  float kreg[32];
  #pragma unroll
  for (int j = 0; j < 32; ++j) kreg[j] = krow[j];
  const float inv_sqrt_dh = 0.0625f;   // 1/sqrt(256)
  for (int b = 0; b < B_; ++b) {
    const float* q = h + ((size_t)b * T_ + t) * D_ + lane * 32;
    float p = 0.f;
    #pragma unroll
    for (int j = 0; j < 32; ++j) p += kreg[j] * q[j];
    p += __shfl_xor(p, 1, 32);
    p += __shfl_xor(p, 2, 32);
    p += __shfl_xor(p, 4, 32);
    if ((lane & 7) == 0) {
      int hh = lane >> 3;
      scores[((size_t)b * H_ + hh) * S_ + s] = p * inv_sqrt_dh;
    }
  }
}

// Top-8 + softmax + weighted gather of V rows. One block per (b,h).
__global__ __launch_bounds__(256) void topk_read_kernel(
    const float* __restrict__ scores, const float* __restrict__ mV,
    float* __restrict__ vt)
{
  const int b  = blockIdx.x >> 2;
  const int hh = blockIdx.x & 3;
  const int tid = threadIdx.x;
  __shared__ float sc[S_];
  __shared__ float rv[256];
  __shared__ int   ri[256];
  __shared__ float topv[TOPK_];
  __shared__ int   topi[TOPK_];
  __shared__ float w[TOPK_];
  const float* srow = scores + ((size_t)b * H_ + hh) * S_;
  for (int i = tid; i < S_; i += 256) sc[i] = srow[i];
  __syncthreads();
  for (int k = 0; k < TOPK_; ++k) {
    float bv = -__builtin_inff(); int bi = 0x7fffffff;
    for (int i = tid; i < S_; i += 256) {
      float v = sc[i];
      if (v > bv || (v == bv && i < bi)) { bv = v; bi = i; }
    }
    rv[tid] = bv; ri[tid] = bi;
    __syncthreads();
    for (int off = 128; off > 0; off >>= 1) {
      if (tid < off) {
        float v2 = rv[tid + off]; int i2 = ri[tid + off];
        if (v2 > rv[tid] || (v2 == rv[tid] && i2 < ri[tid])) { rv[tid] = v2; ri[tid] = i2; }
      }
      __syncthreads();
    }
    if (tid == 0) { topv[k] = rv[0]; topi[k] = ri[0]; sc[ri[0]] = -__builtin_inff(); }
    __syncthreads();
  }
  if (tid == 0) {
    float mx = topv[0], ssum = 0.f;
    for (int k = 0; k < TOPK_; ++k) { w[k] = __expf(topv[k] - mx); ssum += w[k]; }
    for (int k = 0; k < TOPK_; ++k) w[k] /= ssum;
  }
  __syncthreads();
  const int d = tid;
  float a = 0.f;
  #pragma unroll
  for (int k = 0; k < TOPK_; ++k)
    a += w[k] * mV[(size_t)topi[k] * D_ + hh * Dh_ + d];
  vt[(size_t)b * D_ + hh * Dh_ + d] = a;
}

// Per-b write statistics: slot = argmax(sw), surprise gate, lr, decay flag.
__global__ __launch_bounds__(256) void stats_kernel(
    const float* __restrict__ scores, const unsigned char* __restrict__ mask,
    int* __restrict__ slot, float* __restrict__ lr, float* __restrict__ decayf, int t)
{
  const int b = blockIdx.x;
  const int tid = threadIdx.x;
  __shared__ float rv[256];
  __shared__ int   ri[256];
  __shared__ float swmax_s;
  const float* s0 = scores + (size_t)b * H_ * S_;
  float bv = -__builtin_inff(); int bi = 0x7fffffff;
  for (int i = tid; i < S_; i += 256) {
    float sw = 0.5f * (s0[i] + s0[S_ + i] + s0[2 * S_ + i] + s0[3 * S_ + i]);
    if (sw > bv || (sw == bv && i < bi)) { bv = sw; bi = i; }
  }
  rv[tid] = bv; ri[tid] = bi;
  __syncthreads();
  for (int off = 128; off > 0; off >>= 1) {
    if (tid < off) {
      float v2 = rv[tid + off]; int i2 = ri[tid + off];
      if (v2 > rv[tid] || (v2 == rv[tid] && i2 < ri[tid])) { rv[tid] = v2; ri[tid] = i2; }
    }
    __syncthreads();
  }
  if (tid == 0) { swmax_s = rv[0]; slot[b] = ri[0]; }
  __syncthreads();
  const float swmax = swmax_s;
  float se = 0.f;
  for (int i = tid; i < S_; i += 256) {
    float sw = 0.5f * (s0[i] + s0[S_ + i] + s0[2 * S_ + i] + s0[3 * S_ + i]);
    se += __expf(sw - swmax);
  }
  rv[tid] = se; __syncthreads();
  for (int off = 128; off > 0; off >>= 1) { if (tid < off) rv[tid] += rv[tid + off]; __syncthreads(); }
  if (tid == 0) {
    float pmax = 1.0f / rv[0];
    float surprise = 1.0f - pmax;
    float l = (surprise > 0.6f) ? 1.0f : 0.1f;
    l *= (mask[(size_t)b * T_ + t] != 0) ? 1.0f : 0.0f;
    lr[b] = l;
    if (b == 0) {
      bool any = false;
      for (int bb = 0; bb < B_; ++bb) any = any || (mask[(size_t)bb * T_ + t] != 0);
      decayf[0] = any ? 0.9995f : 1.0f;
    }
  }
}

// fraw[b][n] = concat(h_t, v_t)[b] @ fuse_W[:,n]
__global__ __launch_bounds__(128) void fuse_kernel(
    const float* __restrict__ h, const float* __restrict__ vt,
    const float* __restrict__ fuseW, float* __restrict__ fraw, int t)
{
  __shared__ float cat[B_][128];
  const int n = blockIdx.x * 128 + threadIdx.x;
  float acc[B_] = {};
  for (int k0 = 0; k0 < 2 * D_; k0 += 128) {
    const int k = k0 + threadIdx.x;
    #pragma unroll
    for (int b = 0; b < B_; ++b) {
      float c = (k < D_) ? h[((size_t)b * T_ + t) * D_ + k]
                         : vt[(size_t)b * D_ + (k - D_)];
      cat[b][threadIdx.x] = c;
    }
    __syncthreads();
    for (int kk = 0; kk < 128; ++kk) {
      float wv = fuseW[(size_t)(k0 + kk) * D_ + n];
      #pragma unroll
      for (int b = 0; b < B_; ++b) acc[b] += cat[b][kk] * wv;
    }
    __syncthreads();
  }
  #pragma unroll
  for (int b = 0; b < B_; ++b) fraw[(size_t)b * D_ + n] = acc[b];
}

// out[b,t,:] = LN(fraw + fuse_b + h_t) with mln params. One block per b.
__global__ __launch_bounds__(256) void lnout_kernel(
    const float* __restrict__ fraw, const float* __restrict__ fuse_b,
    const float* __restrict__ h, const float* __restrict__ mg,
    const float* __restrict__ mb, float* __restrict__ out, int t)
{
  const int b = blockIdx.x;
  const int tid = threadIdx.x;
  __shared__ float red[256];
  __shared__ float m_s, v_s;
  float s1 = 0.f, s2 = 0.f;
  for (int d = tid; d < D_; d += 256) {
    float z = fraw[(size_t)b * D_ + d] + fuse_b[d] + h[((size_t)b * T_ + t) * D_ + d];
    s1 += z; s2 += z * z;
  }
  red[tid] = s1; __syncthreads();
  for (int off = 128; off > 0; off >>= 1) { if (tid < off) red[tid] += red[tid + off]; __syncthreads(); }
  if (tid == 0) m_s = red[0] * (1.0f / D_);
  __syncthreads();
  red[tid] = s2; __syncthreads();
  for (int off = 128; off > 0; off >>= 1) { if (tid < off) red[tid] += red[tid + off]; __syncthreads(); }
  if (tid == 0) v_s = red[0] * (1.0f / D_) - m_s * m_s;
  __syncthreads();
  const float rs = rsqrtf(v_s + 1e-5f);
  for (int d = tid; d < D_; d += 256) {
    float z = fraw[(size_t)b * D_ + d] + fuse_b[d] + h[((size_t)b * T_ + t) * D_ + d];
    out[((size_t)b * T_ + t) * D_ + d] = (z - m_s) * rs * mg[d] + mb[d];
  }
}

// mem_V *= decay (after the read, before the scatter)
__global__ void decay_kernel(float* __restrict__ mV, const float* __restrict__ decayf) {
  const float d = decayf[0];
  size_t i = (size_t)blockIdx.x * 256 + threadIdx.x;
  mV[i] *= d;
}

// Scatter write matching .at[slot].add semantics (deltas vs pre-update rows).
__global__ __launch_bounds__(256) void scatter_kernel(
    float* __restrict__ mK, float* __restrict__ mV,
    const int* __restrict__ slot, const float* __restrict__ lr,
    const float* __restrict__ h, const float* __restrict__ out, int t)
{
  int d = threadIdx.x;
  for (int rep = 0; rep < D_ / 256; ++rep, d += 256) {
    float kold[B_], vold[B_];
    #pragma unroll
    for (int b = 0; b < B_; ++b) {
      kold[b] = mK[(size_t)slot[b] * D_ + d];
      vold[b] = mV[(size_t)slot[b] * D_ + d];
    }
    #pragma unroll
    for (int b = 0; b < B_; ++b) {
      float l = lr[b];
      mK[(size_t)slot[b] * D_ + d] += l * (h[((size_t)b * T_ + t) * D_ + d]   - kold[b]);
      mV[(size_t)slot[b] * D_ + d] += l * (out[((size_t)b * T_ + t) * D_ + d] - vold[b]);
    }
  }
}

// ---------------------------------------------------------------------------
extern "C" void kernel_launch(void* const* d_in, const int* in_sizes, int n_in,
                              void* d_out, int out_size, void* d_ws, size_t ws_size,
                              hipStream_t stream)
{
  const float* x      = (const float*)d_in[0];
  const unsigned char* wmask = (const unsigned char*)d_in[1];
  const float* W1     = (const float*)d_in[2];
  const float* b1     = (const float*)d_in[3];
  const float* W2     = (const float*)d_in[4];
  const float* b2     = (const float*)d_in[5];
  const float* ln_g   = (const float*)d_in[6];
  const float* ln_b   = (const float*)d_in[7];
  const float* fuse_W = (const float*)d_in[8];
  const float* fuse_b = (const float*)d_in[9];
  const float* mln_g  = (const float*)d_in[10];
  const float* mln_b  = (const float*)d_in[11];
  const float* memK   = (const float*)d_in[12];
  const float* memV   = (const float*)d_in[13];

  char* ws = (char*)d_ws;
  size_t off = 0;
  auto alloc = [&](size_t bytes) -> void* {
    void* p = ws + off;
    off += bytes;
    off = (off + 255) & ~(size_t)255;
    return p;
  };

  const size_t MT = (size_t)B_ * T_;
  float* hbuf   = (float*)alloc(MT * D_ * 4);
  bf16*  hbf    = (bf16*) alloc(MT * D_ * 2);
  bf16*  gbuf   = (bf16*) alloc(MT * 2 * D_ * 2);
  float* ybuf   = (float*)alloc(MT * D_ * 4);
  bf16*  w1t    = (bf16*) alloc((size_t)2 * 2 * D_ * D_ * 2);
  bf16*  w2t    = (bf16*) alloc((size_t)2 * D_ * 2 * D_ * 2);
  float* mK     = (float*)alloc((size_t)S_ * D_ * 4);
  float* mV     = (float*)alloc((size_t)S_ * D_ * 4);
  float* scores = (float*)alloc((size_t)B_ * H_ * S_ * 4);
  float* vt     = (float*)alloc((size_t)B_ * D_ * 4);
  float* fraw   = (float*)alloc((size_t)B_ * D_ * 4);
  int*   slot   = (int*)  alloc(64);
  float* lrbuf  = (float*)alloc(64);
  float* decayf = (float*)alloc(64);
  (void)in_sizes; (void)n_in; (void)out_size; (void)ws_size;

  (void)hipMemcpyAsync(hbuf, x,    MT * D_ * 4,         hipMemcpyDeviceToDevice, stream);
  (void)hipMemcpyAsync(mK,   memK, (size_t)S_ * D_ * 4, hipMemcpyDeviceToDevice, stream);
  (void)hipMemcpyAsync(mV,   memV, (size_t)S_ * D_ * 4, hipMemcpyDeviceToDevice, stream);

  const size_t nW = (size_t)D_ * 2 * D_;
  for (int l = 0; l < 2; ++l) {
    cast_transpose_kernel<<<(int)((nW + 255) / 256), 256, 0, stream>>>(
        W1 + l * nW, w1t + l * nW, D_, 2 * D_);
    cast_transpose_kernel<<<(int)((nW + 255) / 256), 256, 0, stream>>>(
        W2 + l * nW, w2t + l * nW, 2 * D_, D_);
  }
  cast_kernel<<<(int)((MT * D_ + 255) / 256), 256, 0, stream>>>(x, hbf, MT * D_);

  // ---- Backbone: 2 residual MLP blocks via bf16 WMMA ----
  for (int l = 0; l < 2; ++l) {
#if HAVE_TDM
    dim3 g1(2 * D_ / 64, (unsigned)(MT / 64));
    dim3 g2(D_ / 64,     (unsigned)(MT / 64));
#else
    dim3 g1(2 * D_ / 64, (unsigned)(MT / 32));
    dim3 g2(D_ / 64,     (unsigned)(MT / 32));
#endif
    wmma_gemm_kernel<0><<<g1, 256, 0, stream>>>(
        hbf, w1t + l * nW, b1 + l * 2 * D_, gbuf, (int)MT, 2 * D_, D_);
    wmma_gemm_kernel<1><<<g2, 256, 0, stream>>>(
        gbuf, w2t + l * nW, b2 + l * D_, ybuf, (int)MT, D_, 2 * D_);
    lnres_kernel<<<(int)MT, 256, 0, stream>>>(
        ybuf, ln_g + l * D_, ln_b + l * D_, hbuf, hbf);
  }

  // ---- Sequential DualTierMiras scan over T steps ----
  float* outp = (float*)d_out;
  for (int t = 0; t < T_; ++t) {
    scores_kernel   <<<S_ / 8,   256, 0, stream>>>(hbuf, mK, scores, t);
    topk_read_kernel<<<B_ * H_,  256, 0, stream>>>(scores, mV, vt);
    stats_kernel    <<<B_,       256, 0, stream>>>(scores, wmask, slot, lrbuf, decayf, t);
    fuse_kernel     <<<D_ / 128, 128, 0, stream>>>(hbuf, vt, fuse_W, fraw, t);
    lnout_kernel    <<<B_,       256, 0, stream>>>(fraw, fuse_b, hbuf, mln_g, mln_b, outp, t);
    decay_kernel    <<<S_ * D_ / 256, 256, 0, stream>>>(mV, decayf);
    scatter_kernel  <<<1,        256, 0, stream>>>(mK, mV, slot, lrbuf, hbuf, outp, t);
  }
}
